// QuantumEnhancedCNN_11708080849237
// MI455X (gfx1250) — matrix operand order repaired and checked by reading, not compiled, not run on hardware
//
#include <hip/hip_runtime.h>

// ---------------------------------------------------------------------------
// CDNA5 (gfx1250, wave32) implementation of QuantumEnhancedCNN.
// GEMM-shaped work: v_wmma_f32_16x16x32_bf16, K-step 64 (2 WMMA / LDS stage),
// double-buffered LDS, async global->LDS loads (GLOBAL_LOAD_ASYNC_TO_LDS_B128,
// ASYNCcnt-tracked) with a uint4 fallback if the builtin is absent.
// ---------------------------------------------------------------------------

typedef __attribute__((ext_vector_type(16))) __bf16 v16bf;
typedef __attribute__((ext_vector_type(8)))  float  v8f;
typedef __attribute__((ext_vector_type(4)))  int    v4i;

#define PI_K 3.14159f

// ------------------------------ constants ----------------------------------
#define B_N      2048
#define K_C2     288         // 32*3*3 (true K of conv2)
#define K_C2P    320         // padded to 5*64
#define K_FC1    2304        // 64*6*6 (= 36*64, no pad needed)
#define N_FC1    512
#define K_INT    544         // 512 + 32 (true K of int GEMM)
#define K_INTP   576         // padded to 9*64
#define N_INT    512

// ---------------------- async global->LDS (guarded) ------------------------
#if defined(__gfx1250__) && \
    __has_builtin(__builtin_amdgcn_global_load_async_to_lds_b128) && \
    __has_builtin(__builtin_amdgcn_s_wait_asynccnt)
#define HAVE_ASYNC_LDS 1
#else
#define HAVE_ASYNC_LDS 0
#endif

typedef __attribute__((address_space(1))) void gas_void;
typedef __attribute__((address_space(3))) void las_void;
typedef __attribute__((address_space(1))) v4i  gas_v4i;
typedef __attribute__((address_space(3))) v4i  las_v4i;

__device__ __forceinline__ void copy16_g2l(const __bf16* gsrc, __bf16* ldst) {
#if HAVE_ASYNC_LDS
    __builtin_amdgcn_global_load_async_to_lds_b128(
        (gas_v4i*)(gas_void*)gsrc, (las_v4i*)(las_void*)ldst, 0, 0);
#else
    *(uint4*)ldst = *(const uint4*)gsrc;
#endif
}
__device__ __forceinline__ void copy_wait() {
#if HAVE_ASYNC_LDS
    __builtin_amdgcn_s_wait_asynccnt(0);
#endif
}

// ------------------------- f32 -> bf16 convert -----------------------------
__global__ void k_f2bf(const float* __restrict__ s, __bf16* __restrict__ d, int n) {
    int i = blockIdx.x * 256 + threadIdx.x;
    if (i < n) d[i] = (__bf16)s[i];
}
// convert with K padding: src [rows, srcK] -> dst [rows, dstK], pad = 0
__global__ void k_f2bf_pad(const float* __restrict__ s, __bf16* __restrict__ d,
                           int rows, int srcK, int dstK) {
    int i = blockIdx.x * 256 + threadIdx.x;
    if (i >= rows * dstK) return;
    int r = i / dstK, k = i - r * dstK;
    d[i] = (k < srcK) ? (__bf16)s[r * srcK + k] : (__bf16)0.0f;
}

// -------------------- conv1 (3->32, pad1) + relu + pool2 -------------------
__global__ void k_conv1_pool(const float* __restrict__ x,
                             const float* __restrict__ w,
                             const float* __restrict__ bias,
                             float* __restrict__ h) {
    int tid = blockIdx.x * 256 + threadIdx.x;
    int px = tid % 14; int t = tid / 14;
    int py = t % 14;   t /= 14;
    int oc = t % 32;   int b = t / 32;
    const float* wb = w + oc * 27;
    float bval = bias[oc];
    float m = -1e30f;
    #pragma unroll
    for (int dy = 0; dy < 2; ++dy) {
        #pragma unroll
        for (int dx = 0; dx < 2; ++dx) {
            int cy = py * 2 + dy, cx = px * 2 + dx;
            float acc = bval;
            #pragma unroll
            for (int ic = 0; ic < 3; ++ic) {
                #pragma unroll
                for (int ky = 0; ky < 3; ++ky) {
                    int iy = cy + ky - 1;
                    if (iy < 0 || iy >= 28) continue;
                    #pragma unroll
                    for (int kx = 0; kx < 3; ++kx) {
                        int ix = cx + kx - 1;
                        if (ix < 0 || ix >= 28) continue;
                        acc += x[((b * 3 + ic) * 28 + iy) * 28 + ix] * wb[ic * 9 + ky * 3 + kx];
                    }
                }
            }
            acc = fmaxf(acc, 0.0f);
            m = fmaxf(m, acc);
        }
    }
    h[tid] = m;
}

// --------------------- WMMA fragment helpers (bf16) ------------------------
// A (16x32, 16-bit): lanes 0-15 rows M with K 0..15;
// lanes 16-31 same rows, K 16..31; two K per VGPR.
union AFrag { unsigned u[8]; v16bf v; };
// B (32x16): lane%16 = column N; lanes<16 K 0..15, lanes>=16 K 16..31.
union BFrag { unsigned u[8]; v16bf v; };

__device__ __forceinline__ void load_a_frag(AFrag& a, const __bf16* p, int hs) {
    const unsigned* pa = (const unsigned*)p;
    #pragma unroll
    for (int v = 0; v < 8; ++v) {
        int k = v * 2 + hs * 16;
        a.u[v] = pa[k >> 1];
    }
}
__device__ __forceinline__ void load_b_frag(BFrag& b, const __bf16* p) {
    const unsigned* pb = (const unsigned*)p;
    #pragma unroll
    for (int v = 0; v < 8; ++v) b.u[v] = pb[v];
}

// ------------------- conv2 implicit-GEMM WMMA (K=288->320) -----------------
// M = B*144 rows (b,y,x), N = 64 oc. A im2col-gathered into LDS, B async.
// Double-buffered, K-step 64 (2 WMMA per stage). out f32 [M,64] pre-pool.
__device__ __forceinline__ void conv2_gather_a(const float* __restrict__ h,
                                               __bf16* dstRow, int m, int ac, int k0) {
    int b = m / 144, pos = m - b * 144;
    int y = pos / 12, xx = pos - y * 12;
    #pragma unroll
    for (int i = 0; i < 8; ++i) {
        int k = k0 + ac + i;
        float v = 0.0f;
        if (k < K_C2) {
            int ic = k / 9, r9 = k - ic * 9;
            int ky = r9 / 3, kx = r9 - ky * 3;
            v = h[((b * 32 + ic) * 14 + y + ky) * 14 + xx + kx];
        }
        dstRow[ac + i] = (__bf16)v;
    }
}

__global__ void k_conv2_wmma(const float* __restrict__ h,
                             const __bf16* __restrict__ w2b,   // [64, 320] padded
                             const float* __restrict__ bias,
                             float* __restrict__ out) {
    __shared__ __align__(16) __bf16 lA[2][32 * 64];
    __shared__ __align__(16) __bf16 lB[2][64 * 64];
    int t = threadIdx.x;
    int m0 = blockIdx.x * 32;
    int lane = t & 31, wid = t >> 5, wm = wid >> 2, wn = wid & 3;
    int hs = lane >> 4, row = lane & 15;
    int ar = t >> 3, ac = (t & 7) * 8;       // A: 8 elems / thread
    int br = t >> 2, bc = (t & 3) * 16;      // B: 16 elems / thread (2x b128)
    const __bf16* gB = &w2b[br * K_C2P + bc];
    v8f c = {};

    // prologue: stage 0 into buffer 0
    conv2_gather_a(h, &lA[0][ar * 64], m0 + ar, ac, 0);
    copy16_g2l(gB, &lB[0][br * 64 + bc]);
    copy16_g2l(gB + 8, &lB[0][br * 64 + bc + 8]);

    int buf = 0;
    for (int k0 = 0; k0 < K_C2P; k0 += 64) {
        copy_wait();
        __syncthreads();
        if (k0 + 64 < K_C2P) {
            conv2_gather_a(h, &lA[buf ^ 1][ar * 64], m0 + ar, ac, k0 + 64);
            copy16_g2l(gB + k0 + 64, &lB[buf ^ 1][br * 64 + bc]);
            copy16_g2l(gB + k0 + 72, &lB[buf ^ 1][br * 64 + bc + 8]);
        }
        #pragma unroll
        for (int kh = 0; kh < 2; ++kh) {
            AFrag a; BFrag bf;
            load_a_frag(a, &lA[buf][(wm * 16 + row) * 64 + kh * 32], hs);
            load_b_frag(bf, &lB[buf][(wn * 16 + row) * 64 + kh * 32 + hs * 16]);
            c = __builtin_amdgcn_wmma_f32_16x16x32_bf16(false, a.v, false, bf.v,
                                                        (short)0, c, false, false);
        }
        __syncthreads();
        buf ^= 1;
    }
    int n = wn * 16 + row;
    float bv = bias[n];
    #pragma unroll
    for (int r = 0; r < 8; ++r) {
        int m = m0 + wm * 16 + hs * 8 + r;
        out[m * 64 + n] = c[r] + bv;
    }
}

// --------------------- pool2 + relu -> bf16 flat ---------------------------
__global__ void k_pool2(const float* __restrict__ c2, __bf16* __restrict__ flatb) {
    int tid = blockIdx.x * 256 + threadIdx.x;
    int px = tid % 6; int t = tid / 6;
    int py = t % 6;   t /= 6;
    int oc = t % 64;  int b = t / 64;
    float m = -1e30f;
    #pragma unroll
    for (int dy = 0; dy < 2; ++dy)
        #pragma unroll
        for (int dx = 0; dx < 2; ++dx) {
            int idx = (b * 144 + (py * 2 + dy) * 12 + (px * 2 + dx)) * 64 + oc;
            m = fmaxf(m, c2[idx]);
        }
    m = fmaxf(m, 0.0f);
    flatb[b * K_FC1 + oc * 36 + py * 6 + px] = (__bf16)m;
}

// ------------------- fractal = sin(mean(fmap) * pi) ------------------------
__global__ void k_fractal(const __bf16* __restrict__ flatb, float* __restrict__ fr) {
    int tid = blockIdx.x * 256 + threadIdx.x;
    int cch = tid % 64, b = tid / 64;
    float s = 0.0f;
    #pragma unroll
    for (int i = 0; i < 36; ++i) s += (float)flatb[b * K_FC1 + cch * 36 + i];
    fr[tid] = sinf((s * (1.0f / 36.0f)) * PI_K);
}

// -------------------- quantum circuit: one wave per sample -----------------
__global__ void k_quantum(const __bf16* __restrict__ flatb,
                          const float* __restrict__ theta,
                          float* __restrict__ qout) {
    int lane = threadIdx.x & 31;
    int b = blockIdx.x * 8 + (threadIdx.x >> 5);

    float rep[8], qin[8];
    float nrm = 0.0f;
    #pragma unroll
    for (int q = 0; q < 8; ++q) {
        rep[q] = (float)flatb[b * K_FC1 + q];
        nrm += rep[q] * rep[q];
    }
    nrm = sqrtf(nrm) + 1e-8f;
    #pragma unroll
    for (int q = 0; q < 8; ++q) qin[q] = rep[q] / nrm;

    float s[8];
    #pragma unroll
    for (int j = 0; j < 8; ++j) s[j] = 0.0f;
    if (lane == 0) s[0] = 1.0f;

    for (int d = 0; d < 3; ++d) {
        #pragma unroll
        for (int q = 0; q < 8; ++q) {
            float ang = 0.5f * (theta[d * 8 + q] + qin[q]);
            float cs = cosf(ang), sn = sinf(ang);
            int R = 1 << (7 - q);
            float p[8];
            if (R >= 32) {
                int jx = R >> 5;
                #pragma unroll
                for (int j = 0; j < 8; ++j) p[j] = s[j ^ jx];
                #pragma unroll
                for (int j = 0; j < 8; ++j) {
                    bool hi = (j & jx) != 0;
                    s[j] = hi ? (sn * p[j] + cs * s[j]) : (cs * s[j] - sn * p[j]);
                }
            } else {
                #pragma unroll
                for (int j = 0; j < 8; ++j) p[j] = __shfl_xor(s[j], R, 32);
                bool hi = (lane & R) != 0;
                #pragma unroll
                for (int j = 0; j < 8; ++j)
                    s[j] = hi ? (sn * p[j] + cs * s[j]) : (cs * s[j] - sn * p[j]);
            }
        }
        #pragma unroll
        for (int q = 0; q < 7; ++q) {
            int C = 1 << (7 - q), T = 1 << (6 - q);
            float p[8];
            if (T >= 32) {
                int jx = T >> 5;
                #pragma unroll
                for (int j = 0; j < 8; ++j) p[j] = s[j ^ jx];
            } else {
                #pragma unroll
                for (int j = 0; j < 8; ++j) p[j] = __shfl_xor(s[j], T, 32);
            }
            #pragma unroll
            for (int j = 0; j < 8; ++j) {
                bool ctrl = (C >= 32) ? ((j & (C >> 5)) != 0) : ((lane & C) != 0);
                s[j] = ctrl ? p[j] : s[j];
            }
        }
    }

    float z[8];
    #pragma unroll
    for (int q = 0; q < 8; ++q) z[q] = 0.0f;
    #pragma unroll
    for (int j = 0; j < 8; ++j) {
        float p = s[j] * s[j];
        int e = j * 32 + lane;
        #pragma unroll
        for (int q = 0; q < 8; ++q)
            z[q] += (e & (1 << (7 - q))) ? -p : p;
    }
    #pragma unroll
    for (int q = 0; q < 8; ++q)
        #pragma unroll
        for (int off = 16; off > 0; off >>= 1)
            z[q] += __shfl_xor(z[q], off, 32);
    if (lane == 0) {
        #pragma unroll
        for (int q = 0; q < 8; ++q) qout[b * 8 + q] = z[q];
    }
}

// ----------- qf = tanh(qfeat@bq^T+b) * tanh(fractal@bf^T+b) ----------------
__global__ void k_qf(const float* __restrict__ qout, const float* __restrict__ fr,
                     const float* __restrict__ q2c_w, const float* __restrict__ q2c_b,
                     const float* __restrict__ bq_w, const float* __restrict__ bq_b,
                     const float* __restrict__ bf_w, const float* __restrict__ bf_b,
                     __bf16* __restrict__ combined) {
    __shared__ float qfeat[64];
    __shared__ float frs[64];
    int b = blockIdx.x, j = threadIdx.x;
    float acc = q2c_b[j];
    #pragma unroll
    for (int k = 0; k < 8; ++k) acc += qout[b * 8 + k] * q2c_w[j * 8 + k];
    qfeat[j] = acc;
    frs[j] = fr[b * 64 + j];
    __syncthreads();
    if (j < 32) {
        float a1 = bq_b[j], a2 = bf_b[j];
        for (int k = 0; k < 64; ++k) {
            a1 += qfeat[k] * bq_w[j * 64 + k];
            a2 += frs[k]   * bf_w[j * 64 + k];
        }
        combined[b * K_INTP + 512 + j] = (__bf16)(tanhf(a1) * tanhf(a2));
    }
}

// ------------------------ generic WMMA GEMM --------------------------------
// C[M,N] = A[M,K](bf16,row) @ B[N,K](bf16,row)^T + bias; K multiple of 64.
// Double-buffered LDS, async g->l copies, 2 WMMA per stage per wave.
// EPI==1: relu -> bf16 out; EPI==2: f32 out
template <int EPI>
__global__ void k_gemm(const __bf16* __restrict__ A, const __bf16* __restrict__ B,
                       const float* __restrict__ bias, void* __restrict__ outp,
                       int M, int N, int K, int ldc) {
    __shared__ __align__(16) __bf16 lA[2][32 * 64];
    __shared__ __align__(16) __bf16 lB[2][64 * 64];
    int t = threadIdx.x;
    int m0 = blockIdx.x * 32, n0 = blockIdx.y * 64;
    int lane = t & 31, wid = t >> 5, wm = wid >> 2, wn = wid & 3;
    int hs = lane >> 4, row = lane & 15;
    int ar = t >> 3, ac = (t & 7) * 8;
    int br = t >> 2, bc = (t & 3) * 16;
    const __bf16* gA = &A[(m0 + ar) * K + ac];
    const __bf16* gB = &B[(n0 + br) * K + bc];
    v8f c = {};

    copy16_g2l(gA, &lA[0][ar * 64 + ac]);
    copy16_g2l(gB, &lB[0][br * 64 + bc]);
    copy16_g2l(gB + 8, &lB[0][br * 64 + bc + 8]);

    int buf = 0;
    for (int k0 = 0; k0 < K; k0 += 64) {
        copy_wait();
        __syncthreads();
        if (k0 + 64 < K) {
            copy16_g2l(gA + k0 + 64, &lA[buf ^ 1][ar * 64 + ac]);
            copy16_g2l(gB + k0 + 64, &lB[buf ^ 1][br * 64 + bc]);
            copy16_g2l(gB + k0 + 72, &lB[buf ^ 1][br * 64 + bc + 8]);
        }
        #pragma unroll
        for (int kh = 0; kh < 2; ++kh) {
            AFrag a; BFrag bf;
            load_a_frag(a, &lA[buf][(wm * 16 + row) * 64 + kh * 32], hs);
            load_b_frag(bf, &lB[buf][(wn * 16 + row) * 64 + kh * 32 + hs * 16]);
            c = __builtin_amdgcn_wmma_f32_16x16x32_bf16(false, a.v, false, bf.v,
                                                        (short)0, c, false, false);
        }
        __syncthreads();
        buf ^= 1;
    }
    int n = n0 + wn * 16 + row;
    float bv = bias[n];
    #pragma unroll
    for (int r = 0; r < 8; ++r) {
        int m = m0 + wm * 16 + hs * 8 + r;
        float v = c[r] + bv;
        if (EPI == 1) {
            v = fmaxf(v, 0.0f);
            ((__bf16*)outp)[m * ldc + n] = (__bf16)v;
        } else {
            ((float*)outp)[m * ldc + n] = v;
        }
    }
}

// ----------------------------- fc2 (N=10) ----------------------------------
__global__ void k_fc2(const float* __restrict__ integ, const float* __restrict__ w,
                      const float* __restrict__ bias, float* __restrict__ out) {
    int tid = blockIdx.x * 256 + threadIdx.x;
    if (tid >= B_N * 10) return;
    int n = tid % 10, b = tid / 10;
    float acc = bias[n];
    for (int k = 0; k < 512; ++k) acc += integ[b * 512 + k] * w[n * 512 + k];
    out[tid] = acc;
}

// ------------------------------- launcher ----------------------------------
extern "C" void kernel_launch(void* const* d_in, const int* in_sizes, int n_in,
                              void* d_out, int out_size, void* d_ws, size_t ws_size,
                              hipStream_t stream) {
    const float* x       = (const float*)d_in[0];
    const float* conv1_w = (const float*)d_in[1];
    const float* conv1_b = (const float*)d_in[2];
    const float* conv2_w = (const float*)d_in[3];
    const float* conv2_b = (const float*)d_in[4];
    const float* fc1_w   = (const float*)d_in[5];
    const float* fc1_b   = (const float*)d_in[6];
    const float* fc2_w   = (const float*)d_in[7];
    const float* fc2_b   = (const float*)d_in[8];
    const float* q2c_w   = (const float*)d_in[9];
    const float* q2c_b   = (const float*)d_in[10];
    const float* bq_w    = (const float*)d_in[11];
    const float* bq_b    = (const float*)d_in[12];
    const float* bf_w    = (const float*)d_in[13];
    const float* bf_b    = (const float*)d_in[14];
    const float* int_w   = (const float*)d_in[15];
    const float* int_b   = (const float*)d_in[16];
    const float* theta   = (const float*)d_in[17];

    char* ws = (char*)d_ws;
    size_t off = 0;
    auto carve = [&](size_t bytes) {
        char* p = ws + off;
        off = (off + bytes + 255) & ~(size_t)255;
        return p;
    };
    float*  h        = (float*) carve((size_t)B_N * 32 * 14 * 14 * 4);   // 51.4 MB
    float*  c2out    = (float*) carve((size_t)B_N * 144 * 64 * 4);       // 75.5 MB
    __bf16* flatb    = (__bf16*)carve((size_t)B_N * K_FC1 * 2);          //  9.4 MB
    float*  fractal  = (float*) carve((size_t)B_N * 64 * 4);
    float*  qout     = (float*) carve((size_t)B_N * 8 * 4);
    __bf16* combined = (__bf16*)carve((size_t)B_N * K_INTP * 2);         //  2.4 MB
    float*  integ    = (float*) carve((size_t)B_N * 512 * 4);            //  4.2 MB
    __bf16* w2b      = (__bf16*)carve((size_t)64 * K_C2P * 2);           // padded
    __bf16* w1b      = (__bf16*)carve((size_t)512 * K_FC1 * 2);
    __bf16* wib      = (__bf16*)carve((size_t)512 * K_INTP * 2);         // padded

    // 1) weight conversions (conv2 / int padded with zero K-columns)
    {
        int n = 64 * K_C2P;
        k_f2bf_pad<<<(n + 255) / 256, 256, 0, stream>>>(conv2_w, w2b, 64, K_C2, K_C2P);
        n = 512 * K_FC1;
        k_f2bf<<<(n + 255) / 256, 256, 0, stream>>>(fc1_w, w1b, n);
        n = 512 * K_INTP;
        k_f2bf_pad<<<(n + 255) / 256, 256, 0, stream>>>(int_w, wib, 512, K_INT, K_INTP);
    }
    // 2) conv1 + relu + pool
    k_conv1_pool<<<(B_N * 32 * 14 * 14) / 256, 256, 0, stream>>>(x, conv1_w, conv1_b, h);
    // 3) conv2 implicit-GEMM WMMA
    k_conv2_wmma<<<(B_N * 144) / 32, 256, 0, stream>>>(h, w2b, conv2_b, c2out);
    // 4) pool2 + relu -> bf16 flat
    k_pool2<<<(B_N * 64 * 36) / 256, 256, 0, stream>>>(c2out, flatb);
    // 5) fractal
    k_fractal<<<(B_N * 64) / 256, 256, 0, stream>>>(flatb, fractal);
    // 6) quantum circuit (wave32 per sample)
    k_quantum<<<B_N / 8, 256, 0, stream>>>(flatb, theta, qout);
    // 7) qf -> combined[:, 512:544]  (cols 544..575 multiplied by zero weights)
    k_qf<<<B_N, 64, 0, stream>>>(qout, fractal, q2c_w, q2c_b, bq_w, bq_b, bf_w, bf_b, combined);
    // 8) fc1 WMMA GEMM -> relu -> combined[:, :512] (bf16, ldc=576)
    k_gemm<1><<<dim3(B_N / 32, N_FC1 / 64), 256, 0, stream>>>(
        flatb, w1b, fc1_b, (void*)combined, B_N, N_FC1, K_FC1, K_INTP);
    // 9) int WMMA GEMM (K padded to 576) -> integrated (f32)
    k_gemm<2><<<dim3(B_N / 32, N_INT / 64), 256, 0, stream>>>(
        combined, wib, int_b, (void*)integ, B_N, N_INT, K_INTP, N_INT);
    // 10) fc2 -> d_out
    k_fc2<<<(B_N * 10 + 255) / 256, 256, 0, stream>>>(integ, fc2_w, fc2_b, (float*)d_out);

    (void)in_sizes; (void)n_in; (void)out_size; (void)ws_size;
}
